// GraphConvolution_34239479284173
// MI455X (gfx1250) — matrix operand order repaired and checked by reading, not compiled, not run on hardware
//
#include <hip/hip_runtime.h>

#define D_IN  256
#define D_OUT 32

typedef __attribute__((ext_vector_type(16))) __bf16 v16bf;
typedef __attribute__((ext_vector_type(8)))  float  v8f;

__device__ __forceinline__ void split_bf16(float f, __bf16& hi, __bf16& lo) {
  hi = (__bf16)f;                    // RNE truncation to bf16
  lo = (__bf16)(f - (float)hi);      // residual, also bf16
}

// ---------------------------------------------------------------------------
// Kernel 1: pre_sup = x @ W  via v_wmma_f32_16x16x32_bf16 with bf16 splitting.
// One wave handles 2 M-tiles x 2 N-tiles (32 rows x 32 cols), K loop of 8x32.
// N_NODES (100000) is an exact multiple of 16, so no partial M tiles.
// ---------------------------------------------------------------------------
__global__ __launch_bounds__(256)
void gcn_gemm_wmma(const float* __restrict__ x, const float* __restrict__ w,
                   float* __restrict__ pre, int n_mtiles) {
  __shared__ float wlds[D_IN * D_OUT];                 // 32 KB of 320 KB WGP LDS
  for (int i = threadIdx.x; i < D_IN * D_OUT; i += 256) wlds[i] = w[i];
  __syncthreads();

  const int lane = threadIdx.x & 31;
  const int wv   = threadIdx.x >> 5;
  const int job  = blockIdx.x * 8 + wv;                // each job = 2 M tiles
  const int mt0  = job * 2;
  if (mt0 >= n_mtiles) return;                         // wave-uniform exit
  const bool has2 = (mt0 + 1) < n_mtiles;              // wave-uniform

  // ISA 16-bit A 16x32 layout: lanes 0-15 hold K 0..7 & 16..23; lanes 16-31
  // hold K 8..15 & 24..31 for the same row M = lane&15.
  const int m    = lane & 15;
  const int ksel = lane >> 4;                          // 0 or 1
  const int koff = ksel * 8;                           // A-frag K base within 32
  const int kbb  = ksel * 16;                          // unused; B frag also uses koff
  const int col  = lane & 15;                          // B/C column

  const float* xr0 = x + (size_t)(mt0 * 16 + m) * D_IN;
  const float* xr1 = xr0 + (size_t)16 * D_IN;

  v8f acc00 = {}, acc01 = {}, acc10 = {}, acc11 = {};

  for (int k0 = 0; k0 < D_IN; k0 += 32) {
    // --- B fragments (32x16, two N-tiles), hi/lo split, from LDS ---
    v16bf bh0, bl0, bh1, bl1;
#pragma unroll
    for (int j = 0; j < 16; ++j) {
      float f0 = wlds[(k0 + koff + (j & 7) + (j >> 3) * 16) * D_OUT + col];
      float f1 = wlds[(k0 + koff + (j & 7) + (j >> 3) * 16) * D_OUT + col + 16];
      __bf16 h, l;
      split_bf16(f0, h, l); bh0[j] = h; bl0[j] = l;
      split_bf16(f1, h, l); bh1[j] = h; bl1[j] = l;
    }

    // --- A fragment, M-tile 0 ---
    v16bf ah, al;
#pragma unroll
    for (int i = 0; i < 8; ++i) {
      float f0 = xr0[k0 + koff + i];
      float f1 = xr0[k0 + koff + 16 + i];
      __bf16 h, l;
      split_bf16(f0, h, l); ah[i]     = h; al[i]     = l;
      split_bf16(f1, h, l); ah[8 + i] = h; al[8 + i] = l;
    }
    // a*b ~= ah*bh + ah*bl + al*bh  (drop al*bl, ~2^-16 relative)
    acc00 = __builtin_amdgcn_wmma_f32_16x16x32_bf16(false, ah, false, bh0, (short)0, acc00, false, false);
    acc00 = __builtin_amdgcn_wmma_f32_16x16x32_bf16(false, ah, false, bl0, (short)0, acc00, false, false);
    acc00 = __builtin_amdgcn_wmma_f32_16x16x32_bf16(false, al, false, bh0, (short)0, acc00, false, false);
    acc01 = __builtin_amdgcn_wmma_f32_16x16x32_bf16(false, ah, false, bh1, (short)0, acc01, false, false);
    acc01 = __builtin_amdgcn_wmma_f32_16x16x32_bf16(false, ah, false, bl1, (short)0, acc01, false, false);
    acc01 = __builtin_amdgcn_wmma_f32_16x16x32_bf16(false, al, false, bh1, (short)0, acc01, false, false);

    if (has2) {
      // --- A fragment, M-tile 1 ---
      v16bf ah1, al1;
#pragma unroll
      for (int i = 0; i < 8; ++i) {
        float f0 = xr1[k0 + koff + i];
        float f1 = xr1[k0 + koff + 16 + i];
        __bf16 h, l;
        split_bf16(f0, h, l); ah1[i]     = h; al1[i]     = l;
        split_bf16(f1, h, l); ah1[8 + i] = h; al1[8 + i] = l;
      }
      acc10 = __builtin_amdgcn_wmma_f32_16x16x32_bf16(false, ah1, false, bh0, (short)0, acc10, false, false);
      acc10 = __builtin_amdgcn_wmma_f32_16x16x32_bf16(false, ah1, false, bl0, (short)0, acc10, false, false);
      acc10 = __builtin_amdgcn_wmma_f32_16x16x32_bf16(false, al1, false, bh0, (short)0, acc10, false, false);
      acc11 = __builtin_amdgcn_wmma_f32_16x16x32_bf16(false, ah1, false, bh1, (short)0, acc11, false, false);
      acc11 = __builtin_amdgcn_wmma_f32_16x16x32_bf16(false, ah1, false, bl1, (short)0, acc11, false, false);
      acc11 = __builtin_amdgcn_wmma_f32_16x16x32_bf16(false, al1, false, bh1, (short)0, acc11, false, false);
    }
  }

  // C/D layout: VGPR r -> lanes 0-15: M=r, N=lane; lanes 16-31: M=8+r, N=lane-16
  const int rowadd = ksel * 8;
#pragma unroll
  for (int r = 0; r < 8; ++r) {
    int row = mt0 * 16 + rowadd + r;
    pre[(size_t)row * D_OUT + col]      = acc00[r];
    pre[(size_t)row * D_OUT + col + 16] = acc01[r];
  }
  if (has2) {
#pragma unroll
    for (int r = 0; r < 8; ++r) {
      int row = (mt0 + 1) * 16 + rowadd + r;
      pre[(size_t)row * D_OUT + col]      = acc10[r];
      pre[(size_t)row * D_OUT + col + 16] = acc11[r];
    }
  }
}

// ---------------------------------------------------------------------------
// Kernel 2: scatter-add SpMM. pre_sup (12.8 MB) is L2-resident; atomics land
// in L2. 8 threads per edge, each handles a float4 column chunk.
// ---------------------------------------------------------------------------
__global__ __launch_bounds__(256)
void gcn_spmm_scatter(const int* __restrict__ arow, const int* __restrict__ acol,
                      const float* __restrict__ aval, const float* __restrict__ pre,
                      float* __restrict__ out, int E) {
  long idx = (long)blockIdx.x * 256 + threadIdx.x;
  long e   = idx >> 3;
  if (e >= E) return;
  int c4 = (int)(idx & 7) * 4;
  int r  = arow[e];
  int c  = acol[e];
  float v = aval[e];
  const float4 p = *(const float4*)(pre + (size_t)c * D_OUT + c4);
  float* o = out + (size_t)r * D_OUT + c4;
  atomicAdd(o + 0, v * p.x);
  atomicAdd(o + 1, v * p.y);
  atomicAdd(o + 2, v * p.z);
  atomicAdd(o + 3, v * p.w);
}

// ---------------------------------------------------------------------------
// Kernel 3: in-place ReLU on the accumulated output (float4 vectorized).
// ---------------------------------------------------------------------------
__global__ __launch_bounds__(256)
void gcn_relu(float* __restrict__ out, int n4) {
  int i = blockIdx.x * 256 + threadIdx.x;
  if (i < n4) {
    float4* p = (float4*)out + i;
    float4 v = *p;
    v.x = fmaxf(v.x, 0.0f);
    v.y = fmaxf(v.y, 0.0f);
    v.z = fmaxf(v.z, 0.0f);
    v.w = fmaxf(v.w, 0.0f);
    *p = v;
  }
}

extern "C" void kernel_launch(void* const* d_in, const int* in_sizes, int n_in,
                              void* d_out, int out_size, void* d_ws, size_t ws_size,
                              hipStream_t stream) {
  const float* x       = (const float*)d_in[0];
  const int*   adj_row = (const int*)  d_in[1];
  const int*   adj_col = (const int*)  d_in[2];
  const float* adj_val = (const float*)d_in[3];
  const float* weight  = (const float*)d_in[4];

  const int N = in_sizes[0] / D_IN;   // 100000
  const int E = in_sizes[1];          // 1600000
  float* pre = (float*)d_ws;          // N*32*4 = 12.8 MB scratch

  // segment_sum accumulates into d_out -> must start from zero every call
  hipMemsetAsync(d_out, 0, (size_t)out_size * sizeof(float), stream);

  const int n_mtiles = (N + 15) / 16;               // 6250 (exact)
  const int jobs     = (n_mtiles + 1) / 2;          // 2 M-tiles per wave
  const int gblocks  = (jobs + 7) / 8;              // 8 waves per block
  gcn_gemm_wmma<<<gblocks, 256, 0, stream>>>(x, weight, pre, n_mtiles);

  const long sthreads = (long)E * 8;
  const int  sblocks  = (int)((sthreads + 255) / 256);
  gcn_spmm_scatter<<<sblocks, 256, 0, stream>>>(adj_row, adj_col, adj_val, pre,
                                                (float*)d_out, E);

  const int n4 = out_size / 4;                      // 3.2M floats, /4 exact
  gcn_relu<<<(n4 + 255) / 256, 256, 0, stream>>>((float*)d_out, n4);
}
